// DGCNN_partseg_83013127897224
// MI455X (gfx1250) — matrix operand order repaired and checked by reading, not compiled, not run on hardware
//
#include <hip/hip_runtime.h>
#include <hip/hip_bf16.h>

// ---------------------------------------------------------------------------
// DGCNN part-seg forward for MI455X (gfx1250, wave32, WMMA).
// GEMMs: v_wmma_f32_16x16x32_f16, compile-time K-dim (fully unrolled loop),
// pre-packed zero-padded f16 weights (no guards), branchless LDS-staged B
// tiles shared by all 4 waves (address-select gathers, ds_store_b64 packing),
// fragments fetched as 2x global b128 + 2x ds b128. Compile-time store path
// (no exec-mask epilogue except the 50-row logits guard).
// knn/top-k: LDS-tiled VALU with register-resident sorted top-20.
// BN: two-pass (per-channel block reduction, fused apply+leaky[+max]).
// Workspace requirement: ~127 MB.
// ---------------------------------------------------------------------------

typedef __attribute__((ext_vector_type(16))) _Float16 v16h;
typedef __attribute__((ext_vector_type(8)))  _Float16 v8h;
typedef __attribute__((ext_vector_type(4)))  _Float16 v4h;
typedef __attribute__((ext_vector_type(8)))  float    v8f;

#define NPTS 8192
#define KNN  20
#define NKTOT (NPTS * KNN)
#define LRELU(v) ((v) > 0.f ? (v) : 0.2f * (v))

// ---------------------------------------------------------------------------
// point (N,3) -> x0pad (16,N): rows 0-2 = xyz, rows 3-15 = 0
// ---------------------------------------------------------------------------
__global__ void k_x0(const float* __restrict__ point, float* __restrict__ x0) {
  int n = blockIdx.x * blockDim.x + threadIdx.x;
  if (n < NPTS) {
    x0[0 * NPTS + n] = point[n * 3 + 0];
    x0[1 * NPTS + n] = point[n * 3 + 1];
    x0[2 * NPTS + n] = point[n * 3 + 2];
#pragma unroll
    for (int c = 3; c < 16; ++c) x0[c * NPTS + n] = 0.f;
  }
}

// ---------------------------------------------------------------------------
// Pack W (O,Cdim) f32 -> Wh (Opad,Cpad) f16, zero padded
// ---------------------------------------------------------------------------
__global__ void k_packW(const float* __restrict__ W, _Float16* __restrict__ Wh,
                        int O, int Cdim, int Opad, int Cpad) {
  int i = blockIdx.x * blockDim.x + threadIdx.x;
  if (i >= Opad * Cpad) return;
  int o = i / Cpad, c = i - o * Cpad;
  float v = (o < O && c < Cdim) ? W[o * Cdim + c] : 0.f;
  Wh[i] = (_Float16)v;
}

// W1 (64,6) -> (64,32): diff channels at c=0..2, center channels at c=16..18
__global__ void k_packW1(const float* __restrict__ W1, _Float16* __restrict__ Wh) {
  int i = blockIdx.x * blockDim.x + threadIdx.x;
  if (i >= 64 * 32) return;
  int o = i >> 5, c = i & 31;
  float v = 0.f;
  if (c < 3) v = W1[o * 6 + c];
  else if (c >= 16 && c < 19) v = W1[o * 6 + 3 + (c - 16)];
  Wh[i] = (_Float16)v;
}

// ---------------------------------------------------------------------------
// Squared norms per column of feat (C, N)
// ---------------------------------------------------------------------------
__global__ void k_norms(const float* __restrict__ feat, float* __restrict__ xx, int C) {
  int n = blockIdx.x * blockDim.x + threadIdx.x;
  if (n < NPTS) {
    float s = 0.f;
    for (int c = 0; c < C; ++c) { float v = feat[(size_t)c * NPTS + n]; s = fmaf(v, v, s); }
    xx[n] = s;
  }
}

// ---------------------------------------------------------------------------
// knn: per-thread row, LDS-tiled columns, register top-20 (descending neg_dist)
// ---------------------------------------------------------------------------
template <int CDIM>
__global__ void k_knn(const float* __restrict__ feat, const float* __restrict__ xx,
                      int* __restrict__ idx) {
  constexpr int TS = 128;
  __shared__ float sf[CDIM * TS];
  __shared__ float sx[TS];
  const int tid = threadIdx.x;
  const int n = blockIdx.x * blockDim.x + tid;

  float f[CDIM];
#pragma unroll
  for (int c = 0; c < CDIM; ++c) f[c] = feat[(size_t)c * NPTS + n];
  const float xn = xx[n];

  float best[KNN];
  int bidx[KNN];
#pragma unroll
  for (int j = 0; j < KNN; ++j) { best[j] = -3.4e38f; bidx[j] = 0; }

  for (int m0 = 0; m0 < NPTS; m0 += TS) {
    __syncthreads();
    for (int t = tid; t < CDIM * TS; t += blockDim.x) {
      int c = t / TS, mm = t - c * TS;
      sf[t] = feat[(size_t)c * NPTS + m0 + mm];
    }
    if (tid < TS) sx[tid] = xx[m0 + tid];
    if (m0 + TS < NPTS) __builtin_prefetch(feat + m0 + TS, 0, 0); // global_prefetch_b8
    __syncthreads();

    for (int mm = 0; mm < TS; ++mm) {
      float dot = 0.f;
#pragma unroll
      for (int c = 0; c < CDIM; ++c) dot = fmaf(f[c], sf[c * TS + mm], dot);
      float nd = 2.f * dot - xn - sx[mm];
      if (nd > best[KNN - 1]) {
        best[KNN - 1] = nd; bidx[KNN - 1] = m0 + mm;
#pragma unroll
        for (int j = KNN - 1; j > 0; --j) {
          if (best[j] > best[j - 1]) {
            float tv = best[j]; best[j] = best[j - 1]; best[j - 1] = tv;
            int ti = bidx[j]; bidx[j] = bidx[j - 1]; bidx[j - 1] = ti;
          }
        }
      }
    }
  }
#pragma unroll
  for (int j = 0; j < KNN; ++j) idx[n * KNN + j] = bidx[j];
}

// ---------------------------------------------------------------------------
// WMMA GEMM: out(O,Ncols) = Wh(Opad,CPAD) * B(CPAD,Ncols)
// Block = 4 waves, each wave one 16x16 tile (same tileN, tileM = wave).
// Fully unrolled compile-time K loop. B tile staged in LDS (f16, 40-half row
// stride: conflict-free, 16B aligned). Staging: thread -> fixed column
// (tid&15) and 4 consecutive channels -> one ds_store_b64; all gathers are
// branchless (address selects, no exec-mask divergence).
// MODE 0: B[c,j] = B0[c*Ncols + j]
// MODE 1: edge gather from feat B0 (rows, NPTS):
//         c<CHALF:  B0[c,idx[j]] - B0[c, j/KNN] ; else B0[c-CHALF, j/KNN]
// MODE 3: c<1024: emax[c]; else B0[(c-1024)*NPTS + j]
// STORET: out[j*ldOut + m] (logits). GUARD: predicate stores on m < O.
// ---------------------------------------------------------------------------
template <int MODE, int CPAD, int CHALF, int STORET, int GUARD>
__global__ void k_gemm(const _Float16* __restrict__ Wh, const float* __restrict__ B0,
                       const int* __restrict__ nbr, const float* __restrict__ emax,
                       float* __restrict__ out, int O, int Ncols, int ldOut) {
  __shared__ alignas(16) _Float16 sB[16 * 40];

  const int tid = threadIdx.x;
  const int lane = tid & 31;
  const int wave = tid >> 5;
  const int tileM = (blockIdx.y * 4 + wave) * 16;
  const int tileN = blockIdx.x * 16;
  const bool hi = lane >= 16;
  const int l15 = lane & 15;

  // staging assignment: fixed column jloc, 4 consecutive channels from c0
  const int jloc = tid & 15;
  const int c0 = (tid >> 4) * 4;
  const int jg = tileN + jloc;

  int gm = 0, gn = 0;
  if (MODE == 1) { gn = jg / KNN; gm = nbr[jg]; }

  v8f acc = {};
  const _Float16* wrow = Wh + (size_t)(tileM + l15) * CPAD + (hi ? 8 : 0);
  const _Float16* brow = &sB[l15 * 40 + (hi ? 8 : 0)];

#pragma unroll
  for (int k0 = 0; k0 < CPAD; k0 += 32) {
    __syncthreads();
    v4h h4;
#pragma unroll
    for (int u = 0; u < 4; ++u) {
      const int cg = k0 + c0 + u;
      float v;
      if (MODE == 0) {
        v = B0[(size_t)cg * Ncols + jg];
      } else if (MODE == 1) {
        const bool sel = cg < CHALF;
        const int cc = sel ? cg : cg - CHALF;
        const float* base = B0 + (size_t)cc * NPTS;
        const float L1 = base[sel ? gm : gn];   // unconditional, selected index
        const float L2 = base[gn];              // unconditional
        v = sel ? (L1 - L2) : L1;
      } else { // MODE 3: selected pointer, single unconditional load
        const float* ptr = (cg < 1024) ? (emax + cg)
                                       : (B0 + (size_t)(cg - 1024) * NPTS + jg);
        v = *ptr;
      }
      h4[u] = (_Float16)v;
    }
    *(v4h*)(&sB[jloc * 40 + c0]) = h4;          // one ds_store_b64
    __syncthreads();

    // A fragment: 2x 128-bit global loads (immediate offsets, packed f16)
    v8h alo = *(const v8h*)(wrow + k0);
    v8h ahi = *(const v8h*)(wrow + k0 + 16);
    // B fragment: 2x ds_load_b128
    v8h blo = *(const v8h*)(brow);
    v8h bhi = *(const v8h*)(brow + 16);

    v16h a, b;
#pragma unroll
    for (int i = 0; i < 8; ++i) {
      a[i] = alo[i]; a[i + 8] = ahi[i];
      b[i] = blo[i]; b[i + 8] = bhi[i];
    }
    acc = __builtin_amdgcn_wmma_f32_16x16x32_f16(false, a, false, b, (short)0, acc,
                                                 false, false);
  }

  // ---- epilogue: compile-time store path, base pointer + constant steps ----
  const int m0 = tileM + (hi ? 8 : 0);           // first row this lane holds
  if (STORET) {
    // out[j*ldOut + m]: consecutive m per lane
    float* po = out + (size_t)(tileN + l15) * ldOut + m0;
#pragma unroll
    for (int r = 0; r < 8; ++r) {
      if (!GUARD || (m0 + r) < O) po[r] = acc[r];
    }
  } else {
    // out[m*ldOut + j]: coalesced across lanes, row step = ldOut
    float* po = out + (size_t)m0 * ldOut + tileN + l15;
#pragma unroll
    for (int r = 0; r < 8; ++r) {
      if (!GUARD || (m0 + r) < O) po[(size_t)r * ldOut] = acc[r];
    }
  }
}

// ---------------------------------------------------------------------------
// Per-channel sum / sumsq over M elements (one block per channel, no atomics)
// ---------------------------------------------------------------------------
__global__ void k_stats(const float* __restrict__ z, float* __restrict__ stats, size_t M) {
  __shared__ float ss[256], sq[256];
  const size_t base = (size_t)blockIdx.x * M;
  float s = 0.f, q = 0.f;
  for (size_t i = threadIdx.x; i < M; i += 256) {
    float v = z[base + i];
    s += v; q = fmaf(v, v, q);
  }
  ss[threadIdx.x] = s; sq[threadIdx.x] = q;
  __syncthreads();
  for (int o = 128; o > 0; o >>= 1) {
    if ((int)threadIdx.x < o) { ss[threadIdx.x] += ss[threadIdx.x + o]; sq[threadIdx.x] += sq[threadIdx.x + o]; }
    __syncthreads();
  }
  if (threadIdx.x == 0) { stats[2 * blockIdx.x] = ss[0]; stats[2 * blockIdx.x + 1] = sq[0]; }
}

// ---------------------------------------------------------------------------
// In-place BN + leaky over z (C, M)
// ---------------------------------------------------------------------------
__global__ void k_bn(float* __restrict__ z, const float* __restrict__ stats,
                     const float* __restrict__ g, const float* __restrict__ b,
                     int C, int M) {
  size_t i = (size_t)blockIdx.x * blockDim.x + threadIdx.x;
  if (i >= (size_t)C * M) return;
  int c = (int)(i / (size_t)M);
  float invM = 1.f / (float)M;
  float mean = stats[2 * c] * invM;
  float var = stats[2 * c + 1] * invM - mean * mean;
  float inv = rsqrtf(var + 1e-5f);
  float v = g[c] * (z[i] - mean) * inv + b[c];
  z[i] = LRELU(v);
}

// ---------------------------------------------------------------------------
// BN + leaky + max over KNN: z (C, N*KNN) -> xout (C, N)
// ---------------------------------------------------------------------------
__global__ void k_bn_maxk(const float* __restrict__ z, const float* __restrict__ stats,
                          const float* __restrict__ g, const float* __restrict__ b,
                          float* __restrict__ xout, int C) {
  size_t i = (size_t)blockIdx.x * blockDim.x + threadIdx.x;
  if (i >= (size_t)C * NPTS) return;
  int c = (int)(i / (size_t)NPTS);
  int n = (int)(i - (size_t)c * NPTS);
  float invM = 1.f / (float)NKTOT;
  float mean = stats[2 * c] * invM;
  float var = stats[2 * c + 1] * invM - mean * mean;
  float inv = rsqrtf(var + 1e-5f);
  float gc = g[c], bc = b[c];
  float mx = -3.4e38f;
  size_t base = (size_t)c * NKTOT + (size_t)n * KNN;
#pragma unroll
  for (int k = 0; k < KNN; ++k) {
    float v = gc * (z[base + k] - mean) * inv + bc;
    mx = fmaxf(mx, LRELU(v));
  }
  xout[(size_t)c * NPTS + n] = mx;
}

// ---------------------------------------------------------------------------
// BN + leaky + global max over N: z6 (1024, N) -> emax (1024)
// ---------------------------------------------------------------------------
__global__ void k_emax(const float* __restrict__ z, const float* __restrict__ stats,
                       const float* __restrict__ g, const float* __restrict__ b,
                       float* __restrict__ emax) {
  __shared__ float sm[256];
  const int c = blockIdx.x;
  float invM = 1.f / (float)NPTS;
  float mean = stats[2 * c] * invM;
  float var = stats[2 * c + 1] * invM - mean * mean;
  float inv = rsqrtf(var + 1e-5f);
  float gc = g[c], bc = b[c];
  float mx = -3.4e38f;
  for (int n = threadIdx.x; n < NPTS; n += 256) {
    float v = gc * (z[(size_t)c * NPTS + n] - mean) * inv + bc;
    mx = fmaxf(mx, LRELU(v));
  }
  sm[threadIdx.x] = mx;
  __syncthreads();
  for (int o = 128; o > 0; o >>= 1) {
    if ((int)threadIdx.x < o) sm[threadIdx.x] = fmaxf(sm[threadIdx.x], sm[threadIdx.x + o]);
    __syncthreads();
  }
  if (threadIdx.x == 0) emax[c] = sm[0];
}

// ---------------------------------------------------------------------------
// Host orchestration
// ---------------------------------------------------------------------------
extern "C" void kernel_launch(void* const* d_in, const int* in_sizes, int n_in,
                              void* d_out, int out_size, void* d_ws, size_t ws_size,
                              hipStream_t stream) {
  (void)in_sizes; (void)n_in; (void)out_size; (void)ws_size;

  const float* point = (const float*)d_in[0];
  // d_in[1] = class_label (unused by the network)
  const float* W1  = (const float*)d_in[2];
  const float* W2  = (const float*)d_in[3];
  const float* W3  = (const float*)d_in[4];
  const float* W4  = (const float*)d_in[5];
  const float* W5  = (const float*)d_in[6];
  const float* W6  = (const float*)d_in[7];
  const float* W8  = (const float*)d_in[8];
  const float* W9  = (const float*)d_in[9];
  const float* W10 = (const float*)d_in[10];
  const float* W11 = (const float*)d_in[11];
  const float* G1 = (const float*)d_in[12], *B1 = (const float*)d_in[13];
  const float* G2 = (const float*)d_in[14], *B2 = (const float*)d_in[15];
  const float* G3 = (const float*)d_in[16], *B3 = (const float*)d_in[17];
  const float* G4 = (const float*)d_in[18], *B4 = (const float*)d_in[19];
  const float* G5 = (const float*)d_in[20], *B5 = (const float*)d_in[21];
  const float* G6 = (const float*)d_in[22], *B6 = (const float*)d_in[23];
  const float* G8 = (const float*)d_in[24], *B8 = (const float*)d_in[25];
  const float* G9 = (const float*)d_in[26], *B9 = (const float*)d_in[27];
  const float* G10 = (const float*)d_in[28], *B10 = (const float*)d_in[29];

  // ---- workspace layout (~127 MB) ----
  float* p = (float*)d_ws;
  float* x0    = p; p += 16 * NPTS;            // padded (16,N) stage-1 features
  int*   idx   = (int*)p; p += NKTOT;
  float* xx    = p; p += NPTS;
  float* stats = p; p += 2048;
  float* emax  = p; p += 1024;
  float* xcat  = p; p += 192 * NPTS;           // x1 | x2 | x3 contiguous
  float* zA    = p; p += (size_t)64 * NKTOT;
  float* zB    = p; p += (size_t)64 * NKTOT;
  float* z6    = p; p += (size_t)1024 * NPTS;
  // packed f16 weight arena (all sizes multiples of 2048 halves)
  _Float16* wh = (_Float16*)p;
  _Float16* W1h = wh;              wh += 64 * 32;
  _Float16* W2h = wh;              wh += 64 * 64;
  _Float16* W3h = wh;              wh += 64 * 128;
  _Float16* W4h = wh;              wh += 64 * 64;
  _Float16* W5h = wh;              wh += 64 * 128;
  _Float16* W6h = wh;              wh += 1024 * 192;
  _Float16* W8h = wh;              wh += 256 * 1216;
  _Float16* W9h = wh;              wh += 256 * 256;
  _Float16* W10h = wh;             wh += 128 * 256;
  _Float16* W11h = wh;             wh += 64 * 128;    // 50 rows padded to 64
  float* x1 = xcat;
  float* x2 = xcat + 64 * NPTS;
  float* x3 = xcat + 128 * NPTS;
  float* outp = (float*)d_out;

  const int nb = NPTS / 256;
  const dim3 gEdge(NKTOT / 16, 1);
  const dim3 gPt(NPTS / 16, 1);
  #define PACKG(n) (((n) + 255) / 256)

  // ---------- weight packing ----------
  k_packW1<<<PACKG(64 * 32), 256, 0, stream>>>(W1, W1h);
  k_packW<<<PACKG(64 * 64), 256, 0, stream>>>(W2, W2h, 64, 64, 64, 64);
  k_packW<<<PACKG(64 * 128), 256, 0, stream>>>(W3, W3h, 64, 128, 64, 128);
  k_packW<<<PACKG(64 * 64), 256, 0, stream>>>(W4, W4h, 64, 64, 64, 64);
  k_packW<<<PACKG(64 * 128), 256, 0, stream>>>(W5, W5h, 64, 128, 64, 128);
  k_packW<<<PACKG(1024 * 192), 256, 0, stream>>>(W6, W6h, 1024, 192, 1024, 192);
  k_packW<<<PACKG(256 * 1216), 256, 0, stream>>>(W8, W8h, 256, 1216, 256, 1216);
  k_packW<<<PACKG(256 * 256), 256, 0, stream>>>(W9, W9h, 256, 256, 256, 256);
  k_packW<<<PACKG(128 * 256), 256, 0, stream>>>(W10, W10h, 128, 256, 128, 256);
  k_packW<<<PACKG(64 * 128), 256, 0, stream>>>(W11, W11h, 50, 128, 64, 128);

  // ---------- stage 1 ----------
  k_x0<<<nb, 256, 0, stream>>>(point, x0);
  k_norms<<<nb, 256, 0, stream>>>(x0, xx, 3);
  k_knn<3><<<nb, 256, 0, stream>>>(x0, xx, idx);

  k_gemm<1, 32, 16, 0, 0><<<gEdge, 128, 0, stream>>>(W1h, x0, idx, nullptr, zA, 64, NKTOT, NKTOT);
  k_stats<<<64, 256, 0, stream>>>(zA, stats, (size_t)NKTOT);
  k_bn<<<PACKG(64 * (size_t)NKTOT), 256, 0, stream>>>(zA, stats, G1, B1, 64, NKTOT);

  k_gemm<0, 64, 0, 0, 0><<<gEdge, 128, 0, stream>>>(W2h, zA, nullptr, nullptr, zB, 64, NKTOT, NKTOT);
  k_stats<<<64, 256, 0, stream>>>(zB, stats, (size_t)NKTOT);
  k_bn_maxk<<<PACKG(64 * NPTS), 256, 0, stream>>>(zB, stats, G2, B2, x1, 64);

  // ---------- stage 2 ----------
  k_norms<<<nb, 256, 0, stream>>>(x1, xx, 64);
  k_knn<64><<<nb, 256, 0, stream>>>(x1, xx, idx);

  k_gemm<1, 128, 64, 0, 0><<<gEdge, 128, 0, stream>>>(W3h, x1, idx, nullptr, zA, 64, NKTOT, NKTOT);
  k_stats<<<64, 256, 0, stream>>>(zA, stats, (size_t)NKTOT);
  k_bn<<<PACKG(64 * (size_t)NKTOT), 256, 0, stream>>>(zA, stats, G3, B3, 64, NKTOT);

  k_gemm<0, 64, 0, 0, 0><<<gEdge, 128, 0, stream>>>(W4h, zA, nullptr, nullptr, zB, 64, NKTOT, NKTOT);
  k_stats<<<64, 256, 0, stream>>>(zB, stats, (size_t)NKTOT);
  k_bn_maxk<<<PACKG(64 * NPTS), 256, 0, stream>>>(zB, stats, G4, B4, x2, 64);

  // ---------- stage 3 ----------
  k_norms<<<nb, 256, 0, stream>>>(x2, xx, 64);
  k_knn<64><<<nb, 256, 0, stream>>>(x2, xx, idx);

  k_gemm<1, 128, 64, 0, 0><<<gEdge, 128, 0, stream>>>(W5h, x2, idx, nullptr, zA, 64, NKTOT, NKTOT);
  k_stats<<<64, 256, 0, stream>>>(zA, stats, (size_t)NKTOT);
  k_bn_maxk<<<PACKG(64 * NPTS), 256, 0, stream>>>(zA, stats, G5, B5, x3, 64);

  // ---------- global feature ----------
  k_gemm<0, 192, 0, 0, 0><<<dim3(NPTS / 16, 16), 128, 0, stream>>>(W6h, xcat, nullptr, nullptr,
                                                                   z6, 1024, NPTS, NPTS);
  k_stats<<<1024, 256, 0, stream>>>(z6, stats, (size_t)NPTS);
  k_emax<<<1024, 256, 0, stream>>>(z6, stats, G6, B6, emax);

  // ---------- head ----------
  k_gemm<3, 1216, 0, 0, 0><<<dim3(NPTS / 16, 4), 128, 0, stream>>>(W8h, xcat, nullptr, emax,
                                                                   zA, 256, NPTS, NPTS);
  k_stats<<<256, 256, 0, stream>>>(zA, stats, (size_t)NPTS);
  k_bn<<<PACKG(256 * NPTS), 256, 0, stream>>>(zA, stats, G8, B8, 256, NPTS);

  k_gemm<0, 256, 0, 0, 0><<<dim3(NPTS / 16, 4), 128, 0, stream>>>(W9h, zA, nullptr, nullptr,
                                                                  zB, 256, NPTS, NPTS);
  k_stats<<<256, 256, 0, stream>>>(zB, stats, (size_t)NPTS);
  k_bn<<<PACKG(256 * NPTS), 256, 0, stream>>>(zB, stats, G9, B9, 256, NPTS);

  k_gemm<0, 256, 0, 0, 0><<<dim3(NPTS / 16, 2), 128, 0, stream>>>(W10h, zB, nullptr, nullptr,
                                                                  zA, 128, NPTS, NPTS);
  k_stats<<<128, 256, 0, stream>>>(zA, stats, (size_t)NPTS);
  k_bn<<<PACKG(128 * NPTS), 256, 0, stream>>>(zA, stats, G10, B10, 128, NPTS);

  // logits: (50,128) x (128,N) stored transposed as (N,50), guard m<50
  k_gemm<0, 128, 0, 1, 1><<<gPt, 128, 0, stream>>>(W11h, zA, nullptr, nullptr, outp,
                                                   50, NPTS, 50);
}
